// CausalSelfAttention_85615878078724
// MI455X (gfx1250) — matrix operand order repaired and checked by reading, not compiled, not run on hardware
//
#include <hip/hip_runtime.h>
#include <hip/hip_bf16.h>
#include <stdint.h>

// ---------------------------------------------------------------------------
// Types / WMMA helpers (CDNA5 / gfx1250, wave32)
// ---------------------------------------------------------------------------
typedef __attribute__((ext_vector_type(16))) __bf16 v16bf;
typedef __attribute__((ext_vector_type(8)))  float  v8f;

union Frag16 { v16bf v; uint32_t u[8]; };
union Frag8f { v8f  v; float    f[8]; };

__device__ inline uint16_t f2bf(float f) {
  uint32_t u = __float_as_uint(f);
  u += 0x7FFFu + ((u >> 16) & 1u);   // round-to-nearest-even
  return (uint16_t)(u >> 16);
}

// Async global->LDS copy of one 16B chunk per lane (ASYNCcnt-tracked).
// LDS address comes from the VGPR operand; generic shared ptr low 32 bits = LDS offset.
__device__ inline void async_copy_b128(const void* gptr, void* lptr) {
  const uint32_t lds = (uint32_t)(uintptr_t)lptr;
  asm volatile("global_load_async_to_lds_b128 %0, %1, off"
               :: "v"(lds), "v"((unsigned long long)(uintptr_t)gptr)
               : "memory");
}
__device__ inline void wait_async() {
  asm volatile("s_wait_asynccnt 0x0" ::: "memory");
}

// A-matrix 16x32 bf16 fragment from LDS (row-major, stride in u16 elements).
// Per-lane K groups are contiguous -> compiler emits 2x ds_load_b128.
__device__ inline void load_a_frag(const uint16_t* base, int stride, int lane, Frag16& fa) {
  const int m = lane & 15, lh = lane >> 4;
  const uint16_t* rp = base + m * stride;
#pragma unroll
  for (int i = 0; i < 8; ++i) {
    const int k = lh * 16 + 2 * i;
    fa.u[i] = *(const uint32_t*)(rp + k);
  }
}

// B-matrix 32x16 fragment from a K-pair-packed, N-major LDS image:
// swz[n*stride + kp] (u32) = { B[2kp+1][n], B[2kp][n] }.
// Per lane: 8 contiguous dwords -> 2x ds_load_b128, no packing VALU.
__device__ inline void load_b_frag_swz(const uint32_t* base, int stride, int lane, Frag16& fb) {
  const int n = lane & 15, lh = lane >> 4;
  const uint32_t* rp = base + n * stride + lh * 8;
#pragma unroll
  for (int v = 0; v < 8; ++v) fb.u[v] = rp[v];
}

// B-matrix 32x16 where element (k,n) lives at base[n*stride + k] (K contiguous
// in a [row][k] buffer), used for Q*K^T reading K tiles as K^T.
__device__ inline void load_b_frag_kcontig(const uint16_t* base, int stride, int lane, Frag16& fb) {
  const int n = lane & 15, lh = lane >> 4;
  const uint16_t* rp = base + n * stride;
#pragma unroll
  for (int v = 0; v < 8; ++v) {
    const int k0 = 2 * v + 16 * lh;
    fb.u[v] = *(const uint32_t*)(rp + k0);
  }
}

__device__ inline v8f wmma_bf16(const Frag16& a, const Frag16& b, v8f c) {
  return __builtin_amdgcn_wmma_f32_16x16x32_bf16(false, a.v, false, b.v,
                                                 (short)0, c, false, false);
}

// Pack two row-major uint4 (8 bf16 of rows 2kp and 2kp+1) into 8 K-pair u32s
// scattered N-major with given stride.
__device__ inline void swz_store8(uint32_t* dst, int stride, uint4 a, uint4 b) {
  const uint32_t aw[4] = {a.x, a.y, a.z, a.w};
  const uint32_t bw[4] = {b.x, b.y, b.z, b.w};
#pragma unroll
  for (int j = 0; j < 4; ++j) {
    dst[(2 * j)     * stride] = (aw[j] & 0xffffu) | (bw[j] << 16);
    dst[(2 * j + 1) * stride] = (aw[j] >> 16)     | (bw[j] & 0xffff0000u);
  }
}

// ---------------------------------------------------------------------------
// fp32 -> bf16 conversion
// ---------------------------------------------------------------------------
__global__ void cvt_f32_bf16_kernel(const float* __restrict__ in,
                                    uint16_t* __restrict__ out, int n) {
  int i = blockIdx.x * blockDim.x + threadIdx.x;
  if (i < n) out[i] = f2bf(in[i]);
}

// ---------------------------------------------------------------------------
// Shared GEMM mainloop: C_tile(128x128) += A(128xK) * B(Kx128), bf16 -> f32.
// 256 threads = 8 waves, wave grid 2(M)x4(N), each wave 4x2 WMMA accumulators.
// Software-pipelined with DOUBLE-BUFFERED LDS tiles:
//   A tile: async DMA (global_load_async_to_lds_b128) into buffer p^1 while
//           computing on buffer p; one s_wait_asynccnt + barrier per iter.
//   B tile: global loads hoisted above compute, swizzle-stored after compute.
// LDS per buffer: A 128x32 row-major (stride 40 u16);
//                 B swizzled [N=128][Kpair=16] u32, stride 28. 2x = 48KB.
// ---------------------------------------------------------------------------
#define ASTR  40
#define BPSTR 28
#define ATILE (128 * ASTR)
#define BTILE (128 * BPSTR)

__device__ inline void gemm_fill_a_async(const uint16_t* __restrict__ A, int lda,
                                         int kt, uint16_t* As, int tid) {
#pragma unroll
  for (int i = tid; i < 512; i += 256) {
    const int r = i >> 2, c4 = i & 3;
    async_copy_b128(A + (size_t)r * lda + kt + c4 * 8, &As[r * ASTR + c4 * 8]);
  }
}

__device__ inline void gemm_load_b_global(const uint16_t* __restrict__ B, int ldb,
                                          int kt, int tid, uint4& a, uint4& b) {
  const int kpr = tid >> 4, nc = tid & 15;
  const uint16_t* g0 = B + (size_t)(kt + 2 * kpr) * ldb + nc * 8;
  a = *(const uint4*)g0;
  b = *(const uint4*)(g0 + ldb);
}

__device__ inline void gemm_store_b_swz(uint32_t* Bsw, int tid, uint4 a, uint4 b) {
  const int kpr = tid >> 4, nc = tid & 15;
  swz_store8(&Bsw[(nc * 8) * BPSTR + kpr], BPSTR, a, b);
}

__device__ inline void gemm_mainloop(const uint16_t* __restrict__ A,  // + m0*lda
                                     const uint16_t* __restrict__ B,  // + n0
                                     int lda, int ldb, int K,
                                     uint16_t* As, uint32_t* Bsw,  // 2x buffers
                                     Frag8f (&acc)[4][2]) {
  const int tid = threadIdx.x, lane = tid & 31, wave = tid >> 5;
  const int wm = wave >> 2, wn = wave & 3;

  // Prologue: stage K-tile 0 into buffer 0.
  gemm_fill_a_async(A, lda, 0, As, tid);
  {
    uint4 b0{}, b1{};
    gemm_load_b_global(B, ldb, 0, tid, b0, b1);
    gemm_store_b_swz(Bsw, tid, b0, b1);
  }
  wait_async();
  __syncthreads();

  int pb = 0;
  for (int kt = 0; kt < K; kt += 32, pb ^= 1) {
    uint16_t* Asc = As  + pb * ATILE;
    uint32_t* Bsc = Bsw + pb * BTILE;
    const bool more = (kt + 32) < K;

    // Stage K-tile kt+32 into the other buffer while computing this one.
    uint4 b0{}, b1{};
    if (more) {
      gemm_fill_a_async(A, lda, kt + 32, As + (pb ^ 1) * ATILE, tid);
      gemm_load_b_global(B, ldb, kt + 32, tid, b0, b1);
      if (kt + 64 < K)   // hint tile after next (global_prefetch_b8)
        __builtin_prefetch(B + (size_t)(kt + 64 + (tid & 31)) * ldb, 0, 1);
    }

    Frag16 af[4], bfr[2];
#pragma unroll
    for (int i = 0; i < 4; ++i)
      load_a_frag(&Asc[(wm * 64 + i * 16) * ASTR], ASTR, lane, af[i]);
#pragma unroll
    for (int j = 0; j < 2; ++j)
      load_b_frag_swz(&Bsc[(wn * 32 + j * 16) * BPSTR], BPSTR, lane, bfr[j]);
#pragma unroll
    for (int i = 0; i < 4; ++i)
#pragma unroll
      for (int j = 0; j < 2; ++j)
        acc[i][j].v = wmma_bf16(af[i], bfr[j], acc[i][j].v);

    if (more) gemm_store_b_swz(Bsw + (pb ^ 1) * BTILE, tid, b0, b1);
    wait_async();        // A(kt+32) landed (no-op on last iteration)
    __syncthreads();     // publish buffer p^1 to all waves
  }
}

// ---------------------------------------------------------------------------
// Kernel 1: QKV projection + bias + RoPE, scatter to q/k/v [B,H,S,DH] bf16.
// M=8192 (B*S), N=6144 (3*D), K=2048.
// ---------------------------------------------------------------------------
__global__ __launch_bounds__(256) void qkv_rope_kernel(
    const uint16_t* __restrict__ Xb, const uint16_t* __restrict__ Wb,
    const float* __restrict__ bias,
    const float* __restrict__ cosT, const float* __restrict__ sinT,
    uint16_t* __restrict__ qbuf, uint16_t* __restrict__ kbuf,
    uint16_t* __restrict__ vbuf) {
  __shared__ __align__(16) uint16_t As[2 * ATILE];
  __shared__ __align__(16) uint32_t Bsw[2 * BTILE];
  const int tid = threadIdx.x, lane = tid & 31, wave = tid >> 5;
  const int wm = wave >> 2, wn = wave & 3;
  const int m0 = blockIdx.x * 128, n0 = blockIdx.y * 128;

  Frag8f acc[4][2];
#pragma unroll
  for (int i = 0; i < 4; ++i)
#pragma unroll
    for (int j = 0; j < 2; ++j)
#pragma unroll
      for (int e = 0; e < 8; ++e) acc[i][j].f[e] = 0.f;

  gemm_mainloop(Xb + (size_t)m0 * 2048, Wb + n0, 2048, 6144, 2048, As, Bsw, acc);

  const int lh = lane >> 4, ln = lane & 15;
#pragma unroll
  for (int i = 0; i < 4; ++i)
#pragma unroll
    for (int j = 0; j < 2; ++j) {
      const int n = n0 + wn * 32 + j * 16 + ln;
      const int which = n >> 11;            // 0=q 1=k 2=v
      const int rem = n & 2047;
      const int h = rem >> 7, d = rem & 127;
      const float bn = bias[n];
#pragma unroll
      for (int e = 0; e < 8; ++e) {
        const int mg = m0 + wm * 64 + i * 16 + e + 8 * lh;
        const int b = mg >> 11, s = mg & 2047;
        float val = acc[i][j].f[e] + bn;
        float partner = __shfl_xor(val, 1, 32);  // RoPE pair lives in lane^1
        const size_t off = (((size_t)(b * 16 + h)) * 2048 + s) * 128 + d;
        if (which == 2) {
          vbuf[off] = f2bf(val);
        } else {
          const float c  = cosT[s * 64 + (d >> 1)];
          const float sn = sinT[s * 64 + (d >> 1)];
          const float r = ((d & 1) == 0) ? (val * c - partner * sn)
                                         : (partner * sn + val * c);
          if (which == 0) qbuf[off] = f2bf(r); else kbuf[off] = f2bf(r);
        }
      }
    }
}

// ---------------------------------------------------------------------------
// Kernel 2: flash attention (causal). Block = 128 thr = 4 waves.
// Each block: one (b,h), 64 query rows (16 per wave). K/V tiles of 64 keys.
// Q/K tiles: async copies, row-major [row][d] stride 136 u16.
// V tile: K-pair-packed D-major image [d=128][kp=32] u32, stride 36.
// ---------------------------------------------------------------------------
#define QSTR  136
#define PSTR  72
#define VPSTR 36

__global__ __launch_bounds__(128) void flash_attn_kernel(
    const uint16_t* __restrict__ Q, const uint16_t* __restrict__ K,
    const uint16_t* __restrict__ V, uint16_t* __restrict__ O) {
  __shared__ __align__(16) uint16_t Qs[64 * QSTR];
  __shared__ __align__(16) uint16_t Ks[64 * QSTR];
  __shared__ __align__(16) uint32_t Vsw[128 * VPSTR];
  __shared__ __align__(16) uint16_t Ps[4 * 16 * PSTR];

  const int tid = threadIdx.x, lane = tid & 31, wave = tid >> 5;
  const int lh = lane >> 4, ln = lane & 15;
  const int bh = blockIdx.x;               // b*16 + h
  const int b = bh >> 4, h = bh & 15;
  const int qbase = blockIdx.y * 64;
  const int qrow0 = wave * 16;

  const uint16_t* qp = Q + (size_t)bh * 2048 * 128;
  const uint16_t* kp = K + (size_t)bh * 2048 * 128;
  const uint16_t* vp = V + (size_t)bh * 2048 * 128;

  // Q tile 64x128 bf16, async
  for (int i = tid; i < 1024; i += 128) {
    const int r = i >> 4, c4 = i & 15;
    async_copy_b128(qp + (size_t)(qbase + r) * 128 + c4 * 8,
                    &Qs[r * QSTR + c4 * 8]);
  }

  Frag8f o[8];
  float m_i[8], l_i[8];
#pragma unroll
  for (int dt = 0; dt < 8; ++dt)
#pragma unroll
    for (int e = 0; e < 8; ++e) o[dt].f[e] = 0.f;
#pragma unroll
  for (int e = 0; e < 8; ++e) { m_i[e] = -3.0e38f; l_i[e] = 0.f; }

  const float scale = 0.08838834764831845f;  // 1/sqrt(128)
  const int nkt = blockIdx.y + 1;            // causal: only tiles <= diagonal
  uint16_t* Psw = &Ps[wave * 16 * PSTR];

  for (int kt = 0; kt < nkt; ++kt) {
    const int kbase = kt * 64;
    __syncthreads();
    // K tile 64x128 bf16, async, row-major
    for (int i = tid; i < 1024; i += 128) {
      const int r = i >> 4, c4 = i & 15;
      async_copy_b128(kp + (size_t)(kbase + r) * 128 + c4 * 8,
                      &Ks[r * QSTR + c4 * 8]);
    }
    // V tile 64x128 bf16 -> K-pair-packed D-major image
#pragma unroll
    for (int c = 0; c < 4; ++c) {
      const int idx = tid + c * 128;
      const int vkp = idx >> 4, nc = idx & 15;
      const uint16_t* g0 = vp + (size_t)(kbase + 2 * vkp) * 128 + nc * 8;
      uint4 a = *(const uint4*)g0;
      uint4 bb = *(const uint4*)(g0 + 128);
      swz_store8(&Vsw[(nc * 8) * VPSTR + vkp], VPSTR, a, bb);
    }
    wait_async();
    __syncthreads();

    // S tile = Q(16x128) * K^T(128x64), 4 K-chunks of 32
    Frag8f sc[4];
#pragma unroll
    for (int nt = 0; nt < 4; ++nt)
#pragma unroll
      for (int e = 0; e < 8; ++e) sc[nt].f[e] = 0.f;
#pragma unroll
    for (int kc = 0; kc < 4; ++kc) {
      Frag16 qa;
      load_a_frag(&Qs[qrow0 * QSTR + kc * 32], QSTR, lane, qa);
#pragma unroll
      for (int nt = 0; nt < 4; ++nt) {
        Frag16 kf;
        load_b_frag_kcontig(&Ks[(nt * 16) * QSTR + kc * 32], QSTR, lane, kf);
        sc[nt].v = wmma_bf16(qa, kf, sc[nt].v);
      }
    }

    // causal mask + scale
#pragma unroll
    for (int nt = 0; nt < 4; ++nt)
#pragma unroll
      for (int e = 0; e < 8; ++e) {
        const int qg = qbase + qrow0 + e + 8 * lh;
        const int kg = kbase + nt * 16 + ln;
        const float s = sc[nt].f[e] * scale;
        sc[nt].f[e] = (kg <= qg) ? s : -3.0e38f;
      }

    // online softmax (per C-layout row e); reductions stay within lane-halves
#pragma unroll
    for (int e = 0; e < 8; ++e) {
      float rm = fmaxf(fmaxf(sc[0].f[e], sc[1].f[e]),
                       fmaxf(sc[2].f[e], sc[3].f[e]));
      rm = fmaxf(rm, __shfl_xor(rm, 1, 32));
      rm = fmaxf(rm, __shfl_xor(rm, 2, 32));
      rm = fmaxf(rm, __shfl_xor(rm, 4, 32));
      rm = fmaxf(rm, __shfl_xor(rm, 8, 32));
      const float mn = fmaxf(m_i[e], rm);
      const float alpha = __expf(m_i[e] - mn);
      float rs = 0.f;
#pragma unroll
      for (int nt = 0; nt < 4; ++nt) {
        const float p = __expf(sc[nt].f[e] - mn);
        sc[nt].f[e] = p;
        rs += p;
      }
      rs += __shfl_xor(rs, 1, 32);
      rs += __shfl_xor(rs, 2, 32);
      rs += __shfl_xor(rs, 4, 32);
      rs += __shfl_xor(rs, 8, 32);
      l_i[e] = alpha * l_i[e] + rs;
      m_i[e] = mn;
#pragma unroll
      for (int dt = 0; dt < 8; ++dt) o[dt].f[e] *= alpha;
      // stage P in per-wave LDS region (C layout -> row-major)
      const int prow = e + 8 * lh;
#pragma unroll
      for (int nt = 0; nt < 4; ++nt)
        Psw[prow * PSTR + nt * 16 + ln] = f2bf(sc[nt].f[e]);
    }

    // O += P(16x64) * V(64x128), 2 K-chunks of 32 keys
#pragma unroll
    for (int kc = 0; kc < 2; ++kc) {
      Frag16 pa;
      load_a_frag(&Psw[kc * 32], PSTR, lane, pa);
#pragma unroll
      for (int dt = 0; dt < 8; ++dt) {
        Frag16 vf;
        load_b_frag_swz(&Vsw[(dt * 16) * VPSTR + kc * 16], VPSTR, lane, vf);
        o[dt].v = wmma_bf16(pa, vf, o[dt].v);
      }
    }
  }

  // normalize + store attn output bf16 [B,S,D]
#pragma unroll
  for (int dt = 0; dt < 8; ++dt)
#pragma unroll
    for (int e = 0; e < 8; ++e) {
      const int qg = qbase + qrow0 + e + 8 * lh;
      const int d = dt * 16 + ln;
      const float val = o[dt].f[e] / l_i[e];
      O[((size_t)(b * 2048 + qg)) * 2048 + h * 128 + d] = f2bf(val);
    }
}

// ---------------------------------------------------------------------------
// Kernel 3: output projection, M=8192, N=2048, K=2048, fp32 out + bias.
// ---------------------------------------------------------------------------
__global__ __launch_bounds__(256) void out_proj_kernel(
    const uint16_t* __restrict__ Ab, const uint16_t* __restrict__ Wb,
    const float* __restrict__ bias, float* __restrict__ out) {
  __shared__ __align__(16) uint16_t As[2 * ATILE];
  __shared__ __align__(16) uint32_t Bsw[2 * BTILE];
  const int tid = threadIdx.x, lane = tid & 31, wave = tid >> 5;
  const int wm = wave >> 2, wn = wave & 3;
  const int m0 = blockIdx.x * 128, n0 = blockIdx.y * 128;

  Frag8f acc[4][2];
#pragma unroll
  for (int i = 0; i < 4; ++i)
#pragma unroll
    for (int j = 0; j < 2; ++j)
#pragma unroll
      for (int e = 0; e < 8; ++e) acc[i][j].f[e] = 0.f;

  gemm_mainloop(Ab + (size_t)m0 * 2048, Wb + n0, 2048, 2048, 2048, As, Bsw, acc);

  const int lh = lane >> 4, ln = lane & 15;
#pragma unroll
  for (int i = 0; i < 4; ++i)
#pragma unroll
    for (int j = 0; j < 2; ++j) {
      const int n = n0 + wn * 32 + j * 16 + ln;
      const float bn = bias[n];
#pragma unroll
      for (int e = 0; e < 8; ++e) {
        const int mg = m0 + wm * 64 + i * 16 + e + 8 * lh;
        out[(size_t)mg * 2048 + n] = acc[i][j].f[e] + bn;
      }
    }
}

// ---------------------------------------------------------------------------
// Launch
// ---------------------------------------------------------------------------
extern "C" void kernel_launch(void* const* d_in, const int* in_sizes, int n_in,
                              void* d_out, int out_size, void* d_ws, size_t ws_size,
                              hipStream_t stream) {
  const float* x    = (const float*)d_in[0];
  const float* cosT = (const float*)d_in[1];
  const float* sinT = (const float*)d_in[2];
  const float* Wqkv = (const float*)d_in[3];
  const float* bqkv = (const float*)d_in[4];
  const float* Wout = (const float*)d_in[5];
  const float* bout = (const float*)d_in[6];
  float* out = (float*)d_out;

  char* ws = (char*)d_ws;
  size_t off = 0;
  auto carve = [&](size_t bytes) {
    char* p = ws + off;
    off += (bytes + 255) & ~(size_t)255;
    return p;
  };
  uint16_t* xb    = (uint16_t*)carve((size_t)8192 * 2048 * 2);
  uint16_t* wqkvb = (uint16_t*)carve((size_t)2048 * 6144 * 2);
  uint16_t* woutb = (uint16_t*)carve((size_t)2048 * 2048 * 2);
  uint16_t* qb    = (uint16_t*)carve((size_t)64 * 2048 * 128 * 2);
  uint16_t* kb    = (uint16_t*)carve((size_t)64 * 2048 * 128 * 2);
  uint16_t* vb    = (uint16_t*)carve((size_t)64 * 2048 * 128 * 2);
  uint16_t* attnb = (uint16_t*)carve((size_t)8192 * 2048 * 2);

  // fp32 -> bf16 staging
  {
    int n = 8192 * 2048;
    cvt_f32_bf16_kernel<<<(n + 255) / 256, 256, 0, stream>>>(x, xb, n);
    n = 2048 * 6144;
    cvt_f32_bf16_kernel<<<(n + 255) / 256, 256, 0, stream>>>(Wqkv, wqkvb, n);
    n = 2048 * 2048;
    cvt_f32_bf16_kernel<<<(n + 255) / 256, 256, 0, stream>>>(Wout, woutb, n);
  }

  // QKV projection + RoPE   (M/BM=64, N/BN=48)
  qkv_rope_kernel<<<dim3(64, 48), 256, 0, stream>>>(xb, wqkvb, bqkv, cosT, sinT,
                                                    qb, kb, vb);

  // Flash attention: (B*H)=64 x (S/64)=32 blocks, 128 threads
  flash_attn_kernel<<<dim3(64, 32), 128, 0, stream>>>(qb, kb, vb, attnb);

  // Output projection  (64 x 16 tiles)
  out_proj_kernel<<<dim3(64, 16), 256, 0, stream>>>(attnb, woutb, bout, out);
}
